// GatedMLPConcat_56839597195715
// MI455X (gfx1250) — matrix operand order repaired and checked by reading, not compiled, not run on hardware
//
#include <hip/hip_runtime.h>
#include <hip/hip_bf16.h>

// ---------------- problem constants (from reference) ----------------
#define NROWS   8192
#define INDIM   1024
#define DD      512
#define NG      32
#define OUTDIM  1024
#define SDIM    1024          // SM*D straight width
#define H1DIM   1536          // (SM+GM)*D concat width

typedef __attribute__((ext_vector_type(16))) __bf16        v16bf;
typedef __attribute__((ext_vector_type(8)))  float         v8f;
typedef __attribute__((ext_vector_type(4)))  unsigned int  u32x4;

union Frag { v16bf v; u32x4 q[2]; };

struct FragSet {            // one k-step of operands for a 32x64 wave tile
    Frag a0, a1;            // A rows m0..m0+15, m0+16..m0+31
    Frag b0, b1, b2, b3;    // B cols n0+{0,16,32,48}
};

__device__ __forceinline__ void load_set(FragSet& s,
                                         const __bf16* __restrict__ arow0,
                                         const __bf16* __restrict__ arow1,
                                         const __bf16* __restrict__ wrow,
                                         int ldw, int k, int kA, int kB)
{
    s.a0.q[0] = *(const u32x4*)(arow0 + k + kA);
    s.a0.q[1] = *(const u32x4*)(arow0 + k + kA + 8);
    s.a1.q[0] = *(const u32x4*)(arow1 + k + kA);
    s.a1.q[1] = *(const u32x4*)(arow1 + k + kA + 8);
    const __bf16* w = wrow + k + kB;
    s.b0.q[0] = *(const u32x4*)(w);     s.b0.q[1] = *(const u32x4*)(w + 8);
    w += (size_t)16 * ldw;
    s.b1.q[0] = *(const u32x4*)(w);     s.b1.q[1] = *(const u32x4*)(w + 8);
    w += (size_t)16 * ldw;
    s.b2.q[0] = *(const u32x4*)(w);     s.b2.q[1] = *(const u32x4*)(w + 8);
    w += (size_t)16 * ldw;
    s.b3.q[0] = *(const u32x4*)(w);     s.b3.q[1] = *(const u32x4*)(w + 8);
}

#define WMMA_BF16(A_, B_, C_) \
    __builtin_amdgcn_wmma_f32_16x16x32_bf16(false, (A_).v, false, (B_).v, (short)0, (C_), false, false)

__device__ __forceinline__ void mma_set(const FragSet& s, v8f acc[8])
{
    acc[0] = WMMA_BF16(s.a0, s.b0, acc[0]);
    acc[1] = WMMA_BF16(s.a0, s.b1, acc[1]);
    acc[2] = WMMA_BF16(s.a0, s.b2, acc[2]);
    acc[3] = WMMA_BF16(s.a0, s.b3, acc[3]);
    acc[4] = WMMA_BF16(s.a1, s.b0, acc[4]);
    acc[5] = WMMA_BF16(s.a1, s.b1, acc[5]);
    acc[6] = WMMA_BF16(s.a1, s.b2, acc[6]);
    acc[7] = WMMA_BF16(s.a1, s.b3, acc[7]);
}

// ---------------- f32 -> bf16 conversion (8 elems/thread, b128 in/out) --------
__global__ __launch_bounds__(256)
void f32_to_bf16_kernel(const float* __restrict__ in, __bf16* __restrict__ out, int n) {
    int i = (blockIdx.x * blockDim.x + threadIdx.x) * 8;
    if (i < n) {
        union { u32x4 q; float f[4]; } lo, hi;
        lo.q = *(const u32x4*)(in + i);
        hi.q = *(const u32x4*)(in + i + 4);
        union { u32x4 q; __bf16 h[8]; } o;
#pragma unroll
        for (int j = 0; j < 4; ++j) { o.h[j] = (__bf16)lo.f[j]; o.h[4 + j] = (__bf16)hi.f[j]; }
        *(u32x4*)(out + i) = o.q;
    }
}

// ---------------- gate bucketing ----------------
__global__ void zero_counts_kernel(int* counts) {
    if (threadIdx.x < NG) counts[threadIdx.x] = 0;
}

__global__ __launch_bounds__(256)
void bucket_kernel(const int* __restrict__ gate_ids, int* __restrict__ counts,
                   int* __restrict__ rowlist) {
    int i = blockIdx.x * blockDim.x + threadIdx.x;
    if (i < NROWS) {
        int g = gate_ids[i];
        int pos = atomicAdd(&counts[g], 1);
        rowlist[g * NROWS + pos] = i;
    }
}

// ---------------- generic WMMA GEMM: out = [relu](A @ W.T + bias) ----------------
// A: M x K bf16 (row-major, lda), W: Ncols x K bf16 (row-major, ldw == K)
// block = 256 threads = 8 waves (2 M x 4 N); block tile 64x256; wave tile 32x64
// K-loop: ping-pong double buffer so WMMAs overlap the next k-step's loads.
template<bool RELU, bool OUT_F32>
__global__ __launch_bounds__(256)
void wmma_gemm_kernel(const __bf16* __restrict__ A, int lda,
                      const __bf16* __restrict__ W, int ldw, int K,
                      const float*  __restrict__ bias,
                      void* __restrict__ out, int ldo, int col_off)
{
    const int lane = threadIdx.x & 31;
    const int wave = threadIdx.x >> 5;
    const int half = lane >> 4;
    const int l16  = lane & 15;
    const int m0   = blockIdx.y * 64  + (wave >> 2) * 32;
    const int n0   = blockIdx.x * 256 + (wave & 3) * 64;
    const int kA   = half * 16;   // A frag: K = kA..kA+15 (contiguous)
    const int kB   = half * 16;   // B frag: K = kB..kB+15 (contiguous)

    const __bf16* arow0 = A + (size_t)(m0 + l16) * lda;
    const __bf16* arow1 = arow0 + (size_t)16 * lda;
    const __bf16* wrow  = W + (size_t)(n0 + l16) * ldw;

    v8f acc[8];
    v8f zero = {};
#pragma unroll
    for (int t = 0; t < 8; ++t) acc[t] = zero;

    FragSet p, q;
    load_set(p, arow0, arow1, wrow, ldw, 0, kA, kB);
#pragma unroll 1
    for (int k0 = 0; k0 < K; k0 += 64) {
        load_set(q, arow0, arow1, wrow, ldw, k0 + 32, kA, kB);
        mma_set(p, acc);
        int knext = k0 + 64;                 // wrap on the last step: harmless
        if (knext >= K) knext = 0;           // in-bounds load, result unused
        load_set(p, arow0, arow1, wrow, ldw, knext, kA, kB);
        mma_set(q, acc);
    }

#pragma unroll
    for (int mi = 0; mi < 2; ++mi) {
#pragma unroll
        for (int t = 0; t < 4; ++t) {
            int n = n0 + t * 16 + l16;
            float bv = bias ? bias[n] : 0.0f;
#pragma unroll
            for (int r = 0; r < 8; ++r) {
                int m = m0 + mi * 16 + half * 8 + r;
                float v = acc[mi * 4 + t][r] + bv;
                if (RELU) v = v > 0.0f ? v : 0.0f;
                if (OUT_F32) ((float*)out)[(size_t)m * ldo + col_off + n] = v;
                else         ((__bf16*)out)[(size_t)m * ldo + col_off + n] = (__bf16)v;
            }
        }
    }
}

// ---------------- gated WMMA GEMM with row gather/scatter ----------------
// For gate g: rows = rowlist[g*NROWS .. +counts[g]), weights = W1g[g*DD : (g+1)*DD)
// Writes relu(x @ W1g_g.T + b1g_g) into hbuf[:, SDIM:SDIM+DD] at original rows.
__global__ __launch_bounds__(256)
void wmma_gemm_gated_kernel(const __bf16* __restrict__ X,
                            const __bf16* __restrict__ W1g,
                            const float*  __restrict__ b1g,
                            __bf16* __restrict__ hbuf,
                            const int* __restrict__ rowlist,
                            const int* __restrict__ counts)
{
    const int g   = blockIdx.z;
    const int cnt = counts[g];
    const int lane = threadIdx.x & 31;
    const int wave = threadIdx.x >> 5;
    const int half = lane >> 4;
    const int l16  = lane & 15;
    const int m0   = blockIdx.y * 64  + (wave >> 2) * 32;
    if (m0 >= cnt) return;                       // wave-uniform exit
    const int n0   = blockIdx.x * 256 + (wave & 3) * 64;
    const int kA   = half * 16;
    const int kB   = half * 16;

    const int* rl = rowlist + g * NROWS;
    int ml0 = m0 + l16;       if (ml0 >= cnt) ml0 = cnt - 1;   // clamp gathers:
    int ml1 = m0 + 16 + l16;  if (ml1 >= cnt) ml1 = cnt - 1;   // EXEC stays all-1s
    const __bf16* arow0 = X + (size_t)rl[ml0] * INDIM;
    const __bf16* arow1 = X + (size_t)rl[ml1] * INDIM;
    const __bf16* wrow  = W1g + ((size_t)g * DD + n0 + l16) * INDIM;

    v8f acc[8];
    v8f zero = {};
#pragma unroll
    for (int t = 0; t < 8; ++t) acc[t] = zero;

    FragSet p, q;
    load_set(p, arow0, arow1, wrow, INDIM, 0, kA, kB);
#pragma unroll 1
    for (int k0 = 0; k0 < INDIM; k0 += 64) {
        load_set(q, arow0, arow1, wrow, INDIM, k0 + 32, kA, kB);
        mma_set(p, acc);
        int knext = k0 + 64;
        if (knext >= INDIM) knext = 0;
        load_set(p, arow0, arow1, wrow, INDIM, knext, kA, kB);
        mma_set(q, acc);
    }

    // precompute scatter rows for this lane's 16 output rows
    int srow[2][8];
#pragma unroll
    for (int mi = 0; mi < 2; ++mi)
#pragma unroll
        for (int r = 0; r < 8; ++r) {
            int mloc = m0 + mi * 16 + half * 8 + r;
            srow[mi][r] = (mloc < cnt) ? rl[mloc] : -1;
        }

#pragma unroll
    for (int mi = 0; mi < 2; ++mi) {
#pragma unroll
        for (int t = 0; t < 4; ++t) {
            int n = n0 + t * 16 + l16;
            float bv = b1g[g * DD + n];
#pragma unroll
            for (int r = 0; r < 8; ++r) {
                if (srow[mi][r] >= 0) {
                    float v = acc[mi * 4 + t][r] + bv;
                    v = v > 0.0f ? v : 0.0f;
                    hbuf[(size_t)srow[mi][r] * H1DIM + SDIM + n] = (__bf16)v;
                }
            }
        }
    }
}

// ---------------- host-side orchestration ----------------
extern "C" void kernel_launch(void* const* d_in, const int* in_sizes, int n_in,
                              void* d_out, int out_size, void* d_ws, size_t ws_size,
                              hipStream_t stream) {
    const float* x   = (const float*)d_in[0];
    const int*   gid = (const int*)  d_in[1];
    const float* W1s = (const float*)d_in[2];
    const float* b1s = (const float*)d_in[3];
    const float* W1g = (const float*)d_in[4];
    const float* b1g = (const float*)d_in[5];
    const float* W2  = (const float*)d_in[6];
    const float* b2  = (const float*)d_in[7];
    const float* W3  = (const float*)d_in[8];
    float* out = (float*)d_out;

    // workspace layout (all offsets 256B aligned)
    char* ws = (char*)d_ws;
    size_t off = 0;
    __bf16* xb   = (__bf16*)(ws + off); off += (size_t)NROWS * INDIM * 2;        // 16 MB
    __bf16* w1sb = (__bf16*)(ws + off); off += (size_t)SDIM * INDIM * 2;         //  2 MB
    __bf16* w1gb = (__bf16*)(ws + off); off += (size_t)NG * DD * INDIM * 2;      // 32 MB
    __bf16* w2b  = (__bf16*)(ws + off); off += (size_t)DD * H1DIM * 2;           // 1.5 MB
    __bf16* w3b  = (__bf16*)(ws + off); off += (size_t)OUTDIM * DD * 2;          //  1 MB
    __bf16* hbuf = (__bf16*)(ws + off); off += (size_t)NROWS * H1DIM * 2;        // 24 MB
    __bf16* h2   = (__bf16*)(ws + off); off += (size_t)NROWS * DD * 2;           //  8 MB
    int* counts  = (int*)(ws + off);    off += 256;
    int* rowlist = (int*)(ws + off);    off += (size_t)NG * NROWS * 4;           //  1 MB

    auto cvt = [&](const float* src, __bf16* dst, int n) {
        f32_to_bf16_kernel<<<(n / 8 + 255) / 256, 256, 0, stream>>>(src, dst, n);
    };
    cvt(x,   xb,   NROWS * INDIM);
    cvt(W1s, w1sb, SDIM * INDIM);
    cvt(W1g, w1gb, NG * DD * INDIM);
    cvt(W2,  w2b,  DD * H1DIM);
    cvt(W3,  w3b,  OUTDIM * DD);

    zero_counts_kernel<<<1, 32, 0, stream>>>(counts);
    bucket_kernel<<<(NROWS + 255) / 256, 256, 0, stream>>>(gid, counts, rowlist);

    // layer 1 straight: hbuf[:, 0:1024] = relu(x @ W1s.T + b1s)
    wmma_gemm_kernel<true, false>
        <<<dim3(SDIM / 256, NROWS / 64), 256, 0, stream>>>(
            xb, INDIM, w1sb, INDIM, INDIM, b1s, hbuf, H1DIM, 0);

    // layer 1 gated (only the selected gate per row): hbuf[:, 1024:1536]
    wmma_gemm_gated_kernel
        <<<dim3(DD / 256, NROWS / 64, NG), 256, 0, stream>>>(
            xb, w1gb, b1g, hbuf, rowlist, counts);

    // layer 2: h2 = relu(hbuf @ W2.T + b2)
    wmma_gemm_kernel<true, false>
        <<<dim3(DD / 256, NROWS / 64), 256, 0, stream>>>(
            hbuf, H1DIM, w2b, H1DIM, H1DIM, b2, h2, DD, 0);

    // layer 3: out = h2 @ W3.T   (f32 output, no bias, no relu)
    wmma_gemm_kernel<false, true>
        <<<dim3(OUTDIM / 256, NROWS / 64), 256, 0, stream>>>(
            h2, DD, w3b, DD, DD, nullptr, out, OUTDIM, 0);
}
